// GlobalContext_Interaction_14405320311343
// MI455X (gfx1250) — matrix operand order repaired and checked by reading, not compiled, not run on hardware
//
#include <hip/hip_runtime.h>
#include <hip/hip_bf16.h>

// ---------------------------------------------------------------------------
// GCViT global-query window attention, fused for MI455X (gfx1250, wave32).
// All GEMMs run on v_wmma_f32_16x16x32_f16 (f16 in, f32 accumulate).
// ---------------------------------------------------------------------------

typedef __attribute__((ext_vector_type(16))) _Float16 v16h;
typedef __attribute__((ext_vector_type(8)))  _Float16 v8h;
typedef __attribute__((ext_vector_type(4)))  _Float16 v4h;
typedef __attribute__((ext_vector_type(8)))  float    v8f;

#define DIM    512
#define HEADS  8
#define HD     64
#define NTOK   64          // tokens per window
#define BW     512         // windows total
#define QK_SCALE 0.125f    // HD^-0.5
#define XPITCH 520         // padded LDS row stride (halves): 1040B -> 4-bank lane stride

// ---- workspace layout (bytes) --------------------------------------------
static constexpr size_t OFF_WQKV  = 0;                                   // 1024*512 f16
static constexpr size_t OFF_WPROJ = OFF_WQKV  + (size_t)1024*512*2;      // 512*512 f16
static constexpr size_t OFF_BIAS  = OFF_WPROJ + (size_t)512*512*2;       // 8*64*64 f32
static constexpr size_t OFF_K     = OFF_BIAS  + (size_t)8*64*64*4;       // [512][8][64][64] f16 (aliased by O)
static constexpr size_t OFF_V     = OFF_K     + (size_t)BW*HEADS*64*64*2;// [512][8][64(d)][64(m)] f16 (transposed)
// total ~68.8 MB -> resident in 192 MB L2

// ---- WMMA helpers ---------------------------------------------------------
static __device__ __forceinline__ v8f wmma_f16(v16h a, v16h b, v8f c) {
    return __builtin_amdgcn_wmma_f32_16x16x32_f16(false, a, false, b, (short)0, c, false, false);
}

// A-matrix 16x32 f16 fragment. Lane (hi = lane>>4) holds
// K = ks*32 + hi*16 + j   (halves 0..15), matching the contiguous B fragment
static __device__ __forceinline__ v16h a_frag_f32(const float* src, int hi, float scale) {
    v16h a;
    const float* p0 = src + hi * 16;
    const float* p1 = src + hi * 16 + 8;
#pragma unroll
    for (int j = 0; j < 8; ++j) {
        a[j]     = (_Float16)(p0[j] * scale);
        a[j + 8] = (_Float16)(p1[j] * scale);
    }
    return a;
}

static __device__ __forceinline__ v16h a_frag_f16(const _Float16* src, int hi) {
    v16h a;
    const v8h lo = *(const v8h*)(src + hi * 16);
    const v8h hv = *(const v8h*)(src + hi * 16 + 8);
#pragma unroll
    for (int j = 0; j < 8; ++j) { a[j] = lo[j]; a[j + 8] = hv[j]; }
    return a;
}

// B-matrix 32x16 f16 fragment: lane = N, 16 contiguous K-halves per lane.
static __device__ __forceinline__ v16h b_frag_f16(const _Float16* p) {
    v16h b;
    v8h lo = *(const v8h*)p;
    v8h hi = *(const v8h*)(p + 8);
#pragma unroll
    for (int j = 0; j < 8; ++j) { b[j] = lo[j]; b[j + 8] = hi[j]; }
    return b;
}

// ---------------------------------------------------------------------------
// Kernel 0: weight -> f16 conversion + per-head relative-position bias table
// ---------------------------------------------------------------------------
__global__ void gcvit_prep(const float* __restrict__ qkv_w,
                           const float* __restrict__ proj_w,
                           const float* __restrict__ rpb,
                           const int*   __restrict__ rel_idx,
                           char* __restrict__ ws) {
    _Float16* wq = (_Float16*)(ws + OFF_WQKV);
    _Float16* wp = (_Float16*)(ws + OFF_WPROJ);
    float*    bs = (float*)(ws + OFF_BIAS);
    const int NQ = 1024 * 512, NP = 512 * 512, NB = 8 * 64 * 64;
    int stride = gridDim.x * blockDim.x;
    for (int i = blockIdx.x * blockDim.x + threadIdx.x; i < NQ; i += stride)
        wq[i] = (_Float16)qkv_w[i];
    for (int i = blockIdx.x * blockDim.x + threadIdx.x; i < NP; i += stride)
        wp[i] = (_Float16)proj_w[i];
    for (int i = blockIdx.x * blockDim.x + threadIdx.x; i < NB; i += stride) {
        int h = i >> 12, nm = i & 4095;
        bs[i] = rpb[rel_idx[nm] * HEADS + h];
    }
}

// ---------------------------------------------------------------------------
// Kernel 1: KV projection.  grid = 512 windows, 16 waves/WG.
// X is staged ONCE per workgroup as f16 in LDS (padded pitch, conflict-free
// b128 fragment reads); all 16 waves then feed WMMA A-operands from LDS.
// Wave w<8  -> K head w  stored row-major  K[b][h][m][d]
// Wave w>=8 -> V head w-8 stored TRANSPOSED Vt[b][h][d][m]
// ---------------------------------------------------------------------------
__global__ __launch_bounds__(512) void gcvit_kv(const float* __restrict__ x,
                                                const float* __restrict__ qkv_b,
                                                char* __restrict__ ws) {
    __shared__ _Float16 xs[NTOK * XPITCH];   // 65 KB

    const int b      = blockIdx.x;
    const int w      = threadIdx.x >> 5;
    const int l      = threadIdx.x & 31;
    const int lane15 = l & 15;
    const int hi     = l >> 4;
    const int c0     = w * 64;

    const _Float16* wq   = (const _Float16*)(ws + OFF_WQKV);
    _Float16*       kbuf = (_Float16*)(ws + OFF_K);
    _Float16*       vbuf = (_Float16*)(ws + OFF_V);
    const float*    xb   = x + (size_t)b * NTOK * DIM;

    // ---- cooperative stage: X (64x512 f32) -> f16 LDS, one pass ----
    {
        const float4* xv = (const float4*)xb;        // 8192 float4
#pragma unroll
        for (int it = 0; it < 16; ++it) {
            int i4  = threadIdx.x + it * 512;
            float4 f = xv[i4];
            int e   = i4 * 4;
            int row = e >> 9;
            int col = e & 511;
            v4h hq;
            hq[0] = (_Float16)f.x; hq[1] = (_Float16)f.y;
            hq[2] = (_Float16)f.z; hq[3] = (_Float16)f.w;
            *(v4h*)&xs[row * XPITCH + col] = hq;
        }
    }
    __syncthreads();

    v8f zero = {};
    v8f acc[4][4];
#pragma unroll
    for (int mt = 0; mt < 4; ++mt)
#pragma unroll
        for (int nt = 0; nt < 4; ++nt) acc[mt][nt] = zero;

    for (int ks = 0; ks < 16; ++ks) {          // K = 512 = 16 * 32
        v16h bf[4];
#pragma unroll
        for (int nt = 0; nt < 4; ++nt) {
            int c = c0 + nt * 16 + lane15;
            bf[nt] = b_frag_f16(wq + (size_t)c * 512 + ks * 32 + hi * 16);
        }
#pragma unroll
        for (int mt = 0; mt < 4; ++mt) {
            int row = mt * 16 + lane15;
            v16h af = a_frag_f16(xs + row * XPITCH + ks * 32, hi);
#pragma unroll
            for (int nt = 0; nt < 4; ++nt)
                acc[mt][nt] = wmma_f16(af, bf[nt], acc[mt][nt]);
        }
    }

    float bq[4];
#pragma unroll
    for (int nt = 0; nt < 4; ++nt) bq[nt] = qkv_b[c0 + nt * 16 + lane15];

#pragma unroll
    for (int mt = 0; mt < 4; ++mt)
#pragma unroll
        for (int nt = 0; nt < 4; ++nt)
#pragma unroll
            for (int r = 0; r < 8; ++r) {
                int m = mt * 16 + r + hi * 8;      // token
                int c = nt * 16 + lane15;          // channel in [0,64)
                float v = acc[mt][nt][r] + bq[nt];
                if (w < 8)
                    kbuf[(((size_t)b * 8 + w) * 64 + m) * 64 + c] = (_Float16)v;
                else
                    vbuf[(((size_t)b * 8 + (w - 8)) * 64 + c) * 64 + m] = (_Float16)v;
            }
}

// ---------------------------------------------------------------------------
// Kernel 2: attention. grid = 512 windows, 8 waves/WG (one head per wave).
// QK^T -> +bias -> register softmax -> P (via LDS relayout) -> P@V.
// O (f16) overwrites the K region after a barrier (K fully consumed).
// ---------------------------------------------------------------------------
__global__ __launch_bounds__(256) void gcvit_attn(const float* __restrict__ q_global,
                                                  char* __restrict__ ws) {
    __shared__ _Float16 lds_p[HEADS * 64 * 64];   // 64 KB

    const int b      = blockIdx.x;
    const int h      = threadIdx.x >> 5;
    const int l      = threadIdx.x & 31;
    const int lane15 = l & 15;
    const int hi     = l >> 4;

    const float*    bias = (const float*)(ws + OFF_BIAS) + (size_t)h * 4096;
    const _Float16* kbuf = (const _Float16*)(ws + OFF_K) + ((size_t)b * 8 + h) * 4096;
    const _Float16* vbuf = (const _Float16*)(ws + OFF_V) + ((size_t)b * 8 + h) * 4096;
    const float*    qsrc = q_global + ((size_t)(b >> 6) * 8 + h) * 4096;

    // Q A-fragments (scale folded in), reused across all N-tiles
    v16h qa[4][2];
#pragma unroll
    for (int mt = 0; mt < 4; ++mt)
#pragma unroll
        for (int ks = 0; ks < 2; ++ks) {
            int row = mt * 16 + lane15;
            qa[mt][ks] = a_frag_f32(qsrc + (size_t)row * 64 + ks * 32, hi, QK_SCALE);
        }

    v8f zero = {};
    v8f acc[4][4];
#pragma unroll
    for (int mt = 0; mt < 4; ++mt)
#pragma unroll
        for (int nt = 0; nt < 4; ++nt) acc[mt][nt] = zero;

    // attn = Q @ K^T   (B fragment lane = key token m, halves = d)
    for (int ks = 0; ks < 2; ++ks) {
        v16h bf[4];
#pragma unroll
        for (int nt = 0; nt < 4; ++nt) {
            int m = nt * 16 + lane15;
            bf[nt] = b_frag_f16(kbuf + (size_t)m * 64 + ks * 32 + hi * 16);
        }
#pragma unroll
        for (int mt = 0; mt < 4; ++mt)
#pragma unroll
            for (int nt = 0; nt < 4; ++nt)
                acc[mt][nt] = wmma_f16(qa[mt][ks], bf[nt], acc[mt][nt]);
    }

    // + relative position bias
#pragma unroll
    for (int mt = 0; mt < 4; ++mt)
#pragma unroll
        for (int nt = 0; nt < 4; ++nt)
#pragma unroll
            for (int r = 0; r < 8; ++r) {
                int n = mt * 16 + r + hi * 8;
                int m = nt * 16 + lane15;
                acc[mt][nt][r] += bias[n * 64 + m];
            }

    // softmax over rows; row lives in one VGPR slot across a 16-lane half
#pragma unroll
    for (int mt = 0; mt < 4; ++mt)
#pragma unroll
        for (int r = 0; r < 8; ++r) {
            float mx = -3.4e38f;
#pragma unroll
            for (int nt = 0; nt < 4; ++nt) mx = fmaxf(mx, acc[mt][nt][r]);
#pragma unroll
            for (int d = 1; d < 16; d <<= 1) mx = fmaxf(mx, __shfl_xor(mx, d, 32));
            float s = 0.f;
#pragma unroll
            for (int nt = 0; nt < 4; ++nt) {
                float e = __expf(acc[mt][nt][r] - mx);
                acc[mt][nt][r] = e;
                s += e;
            }
#pragma unroll
            for (int d = 1; d < 16; d <<= 1) s += __shfl_xor(s, d, 32);
            float inv = 1.0f / s;
#pragma unroll
            for (int nt = 0; nt < 4; ++nt) acc[mt][nt][r] *= inv;
        }

    // P -> LDS (C-layout -> A-layout relayout), per-wave private region
    _Float16* pl = lds_p + h * 4096;
#pragma unroll
    for (int mt = 0; mt < 4; ++mt)
#pragma unroll
        for (int nt = 0; nt < 4; ++nt)
#pragma unroll
            for (int r = 0; r < 8; ++r) {
                int n = mt * 16 + r + hi * 8;
                int m = nt * 16 + lane15;
                pl[n * 64 + m] = (_Float16)acc[mt][nt][r];
            }

    // O = P @ V   (B fragment lane = d, halves = m; Vt is [d][m] row-major)
    v8f oacc[4][4];
#pragma unroll
    for (int mt = 0; mt < 4; ++mt)
#pragma unroll
        for (int nt = 0; nt < 4; ++nt) oacc[mt][nt] = zero;

    for (int ks = 0; ks < 2; ++ks) {
        v16h bf[4];
#pragma unroll
        for (int nt = 0; nt < 4; ++nt) {
            int d = nt * 16 + lane15;
            bf[nt] = b_frag_f16(vbuf + (size_t)d * 64 + ks * 32 + hi * 16);
        }
#pragma unroll
        for (int mt = 0; mt < 4; ++mt) {
            v16h pa = a_frag_f16(pl + (size_t)(mt * 16 + lane15) * 64 + ks * 32, hi);
#pragma unroll
            for (int nt = 0; nt < 4; ++nt)
                oacc[mt][nt] = wmma_f16(pa, bf[nt], oacc[mt][nt]);
        }
    }

    __syncthreads();   // all K reads complete before O overwrites the K region

    _Float16* obuf = (_Float16*)(ws + OFF_K);  // O[b][n][512], aliases K
#pragma unroll
    for (int mt = 0; mt < 4; ++mt)
#pragma unroll
        for (int nt = 0; nt < 4; ++nt)
#pragma unroll
            for (int r = 0; r < 8; ++r) {
                int n = mt * 16 + r + hi * 8;
                int d = nt * 16 + lane15;
                obuf[((size_t)b * 64 + n) * 512 + h * 64 + d] = (_Float16)oacc[mt][nt][r];
            }
}

// ---------------------------------------------------------------------------
// Kernel 3: output projection. grid = 512 windows, 8 waves/WG (64 cols each).
// ---------------------------------------------------------------------------
__global__ __launch_bounds__(256) void gcvit_proj(const float* __restrict__ proj_b,
                                                  const char* __restrict__ ws,
                                                  float* __restrict__ out) {
    const int b      = blockIdx.x;
    const int w      = threadIdx.x >> 5;
    const int l      = threadIdx.x & 31;
    const int lane15 = l & 15;
    const int hi     = l >> 4;
    const int c0     = w * 64;

    const _Float16* obuf = (const _Float16*)(ws + OFF_K) + (size_t)b * 64 * 512;
    const _Float16* wp   = (const _Float16*)(ws + OFF_WPROJ);

    v8f zero = {};
    v8f acc[4][4];
#pragma unroll
    for (int mt = 0; mt < 4; ++mt)
#pragma unroll
        for (int nt = 0; nt < 4; ++nt) acc[mt][nt] = zero;

    for (int ks = 0; ks < 16; ++ks) {
        v16h bf[4];
#pragma unroll
        for (int nt = 0; nt < 4; ++nt) {
            int c = c0 + nt * 16 + lane15;
            bf[nt] = b_frag_f16(wp + (size_t)c * 512 + ks * 32 + hi * 16);
        }
#pragma unroll
        for (int mt = 0; mt < 4; ++mt) {
            v16h af = a_frag_f16(obuf + (size_t)(mt * 16 + lane15) * 512 + ks * 32, hi);
#pragma unroll
            for (int nt = 0; nt < 4; ++nt)
                acc[mt][nt] = wmma_f16(af, bf[nt], acc[mt][nt]);
        }
    }

    float pb[4];
#pragma unroll
    for (int nt = 0; nt < 4; ++nt) pb[nt] = proj_b[c0 + nt * 16 + lane15];

#pragma unroll
    for (int mt = 0; mt < 4; ++mt)
#pragma unroll
        for (int nt = 0; nt < 4; ++nt)
#pragma unroll
            for (int r = 0; r < 8; ++r) {
                int n = mt * 16 + r + hi * 8;
                int c = c0 + nt * 16 + lane15;
                out[((size_t)b * 64 + n) * 512 + c] = acc[mt][nt][r] + pb[nt];
            }
}

// ---------------------------------------------------------------------------
extern "C" void kernel_launch(void* const* d_in, const int* in_sizes, int n_in,
                              void* d_out, int out_size, void* d_ws, size_t ws_size,
                              hipStream_t stream) {
    const float* x        = (const float*)d_in[0];
    const float* q_global = (const float*)d_in[1];
    const float* qkv_w    = (const float*)d_in[2];
    const float* qkv_b    = (const float*)d_in[3];
    const float* proj_w   = (const float*)d_in[4];
    const float* proj_b   = (const float*)d_in[5];
    const float* rpb      = (const float*)d_in[6];
    const int*   rel_idx  = (const int*)d_in[7];
    float*       out      = (float*)d_out;
    char*        ws       = (char*)d_ws;

    (void)in_sizes; (void)n_in; (void)out_size; (void)ws_size;

    gcvit_prep<<<512, 256, 0, stream>>>(qkv_w, proj_w, rpb, rel_idx, ws);
    gcvit_kv  <<<BW, 512, 0, stream>>>(x, qkv_b, ws);
    gcvit_attn<<<BW, 256, 0, stream>>>(q_global, ws);
    gcvit_proj<<<BW, 256, 0, stream>>>(proj_b, ws, out);
}
